// FiraCopyNet_56143812493970
// MI455X (gfx1250) — matrix operand order repaired and checked by reading, not compiled, not run on hardware
//
#include <hip/hip_runtime.h>
#include <hip/hip_bf16.h>

typedef __attribute__((ext_vector_type(16))) _Float16 v16h;
typedef __attribute__((ext_vector_type(8)))  _Float16 v8h;
typedef __attribute__((ext_vector_type(8)))  float    v8f;

#define B_  2
#define S_  256
#define T_  128
#define ES_ 512   // == ET
#define H_  1024

// ---------------------------------------------------------------------------
// One-shot f32 -> f16 conversion, 8 elements / thread (b128 in, b128 out).
// ---------------------------------------------------------------------------
__global__ __launch_bounds__(256)
void cvt_f32_f16(const float* __restrict__ in, _Float16* __restrict__ out, int n8)
{
    const int i = blockIdx.x * 256 + threadIdx.x;
    if (i >= n8) return;
    const float4* p = (const float4*)in;
    const float4 a = p[2 * i], b = p[2 * i + 1];
    v8h o;
    o[0] = (_Float16)a.x; o[1] = (_Float16)a.y; o[2] = (_Float16)a.z; o[3] = (_Float16)a.w;
    o[4] = (_Float16)b.x; o[5] = (_Float16)b.y; o[6] = (_Float16)b.z; o[7] = (_Float16)b.w;
    *(v8h*)(out + 8 * (size_t)i) = o;
}

// ---------------------------------------------------------------------------
// C[M,1024] = Xh[M,512] * Wh[1024,512]^T  via v_wmma_f32_16x16x32_f16.
// One wave per 16x64 output strip: A fragment reused across 4 WMMAs.
// Fragment layouts used by this kernel (wave32):
//   A 16x32: lane holds row M=l15; half 0 -> K 0..15, half 1 -> K 16..31 (contiguous)
//   B 32x16: lane holds col N=l15; half 0 -> K 0..15, half 1 -> K 16..31 (contiguous)
//   D:       VGPR r -> M = 8*half + r, N = l15
// ---------------------------------------------------------------------------
__global__ __launch_bounds__(32)
void gemm_nt_wmma_f16(const _Float16* __restrict__ Xh, const _Float16* __restrict__ Wh,
                      float* __restrict__ C)
{
    const int lane = threadIdx.x;          // block == one wave
    const int n0   = blockIdx.x * 64;
    const int m0   = blockIdx.y * 16;
    const int half = (lane >> 4) & 1;
    const int l15  = lane & 15;

    const _Float16* xrow = Xh + (size_t)(m0 + l15) * ES_ + (half ? 16 : 0);
    const int bkb = half ? 16 : 0;
    const _Float16* w0 = Wh + (size_t)(n0 +  0 + l15) * ES_ + bkb;
    const _Float16* w1 = Wh + (size_t)(n0 + 16 + l15) * ES_ + bkb;
    const _Float16* w2 = Wh + (size_t)(n0 + 32 + l15) * ES_ + bkb;
    const _Float16* w3 = Wh + (size_t)(n0 + 48 + l15) * ES_ + bkb;

    v8f acc0 = {}, acc1 = {}, acc2 = {}, acc3 = {};

#define CAT16(lo, hi) __builtin_shufflevector(lo, hi, 0,1,2,3,4,5,6,7,8,9,10,11,12,13,14,15)

    #pragma unroll 4
    for (int k0 = 0; k0 < ES_; k0 += 32) {
        const v8h alo = *(const v8h*)(xrow + k0);
        const v8h ahi = *(const v8h*)(xrow + k0 + 8);
        const v16h a = CAT16(alo, ahi);

        const v16h b0 = CAT16(*(const v8h*)(w0 + k0), *(const v8h*)(w0 + k0 + 8));
        const v16h b1 = CAT16(*(const v8h*)(w1 + k0), *(const v8h*)(w1 + k0 + 8));
        const v16h b2 = CAT16(*(const v8h*)(w2 + k0), *(const v8h*)(w2 + k0 + 8));
        const v16h b3 = CAT16(*(const v8h*)(w3 + k0), *(const v8h*)(w3 + k0 + 8));

        acc0 = __builtin_amdgcn_wmma_f32_16x16x32_f16(false, a, false, b0, (short)0, acc0, false, false);
        acc1 = __builtin_amdgcn_wmma_f32_16x16x32_f16(false, a, false, b1, (short)0, acc1, false, false);
        acc2 = __builtin_amdgcn_wmma_f32_16x16x32_f16(false, a, false, b2, (short)0, acc2, false, false);
        acc3 = __builtin_amdgcn_wmma_f32_16x16x32_f16(false, a, false, b3, (short)0, acc3, false, false);
    }
#undef CAT16

    float* crow = C + (size_t)(m0 + half * 8) * H_ + n0 + l15;
    #pragma unroll
    for (int r = 0; r < 8; ++r) {
        crow[(size_t)r * H_ +  0] = acc0[r];
        crow[(size_t)r * H_ + 16] = acc1[r];
        crow[(size_t)r * H_ + 32] = acc2[r];
        crow[(size_t)r * H_ + 48] = acc3[r];
    }
}

// ---------------------------------------------------------------------------
// Fused additive attention: cp[b,t,s] = sum_h tanh(sL[b,s,h]+tL[b,t,h])*Wr[h]+br
// One block per (b,t); tL row + Wr in LDS (float4); wave32 shuffle reduction.
// Never materializes [B,T,S,H] (saves ~0.5 GB HBM traffic; sL stays in L2).
// ---------------------------------------------------------------------------
__global__ __launch_bounds__(256)
void attn_tanh_reduce(const float* __restrict__ sL, const float* __restrict__ tL,
                      const float* __restrict__ Wr, const float* __restrict__ br,
                      const unsigned char* __restrict__ mask,
                      float* __restrict__ cp)
{
    __shared__ float4 tls[H_ / 4];
    __shared__ float4 wrs[H_ / 4];

    const int bt  = blockIdx.x;        // b*T + t
    const int b   = bt / T_;
    const int tid = threadIdx.x;

    for (int i = tid; i < H_ / 4; i += 256) {
        tls[i] = ((const float4*)(tL + (size_t)bt * H_))[i];
        wrs[i] = ((const float4*)Wr)[i];
    }
    __syncthreads();

    const float brv  = br[0];
    const int   wave = tid >> 5;
    const int   lane = tid & 31;

    for (int s = wave; s < S_; s += 8) {
        const float4* sl4 = (const float4*)(sL + (size_t)(b * S_ + s) * H_);
        float acc = 0.f;
        #pragma unroll 2
        for (int h4 = lane; h4 < H_ / 4; h4 += 32) {
            const float4 sv = sl4[h4];
            const float4 tv = tls[h4];
            const float4 wv = wrs[h4];
            acc += tanhf(sv.x + tv.x) * wv.x;
            acc += tanhf(sv.y + tv.y) * wv.y;
            acc += tanhf(sv.z + tv.z) * wv.z;
            acc += tanhf(sv.w + tv.w) * wv.w;
        }
        #pragma unroll
        for (int off = 16; off > 0; off >>= 1)
            acc += __shfl_xor(acc, off, 32);
        if (lane == 0)
            cp[(size_t)bt * S_ + s] =
                mask[b * S_ + s] ? (acc + brv) : -__builtin_inff();
    }
}

// ---------------------------------------------------------------------------
// Gate: softmax(target[b,t,:] @ Wg^T + bg) over 2 classes. One wave per (b,t).
// ---------------------------------------------------------------------------
__global__ __launch_bounds__(32)
void gate_softmax(const float* __restrict__ target, const float* __restrict__ Wg,
                  const float* __restrict__ bg, float* __restrict__ gate)
{
    const int bt   = blockIdx.x;
    const int lane = threadIdx.x;
    const float* tg = target + (size_t)bt * ES_;

    float g0 = 0.f, g1 = 0.f;
    for (int e = lane; e < ES_; e += 32) {
        const float v = tg[e];
        g0 += v * Wg[e];
        g1 += v * Wg[ES_ + e];
    }
    #pragma unroll
    for (int off = 16; off > 0; off >>= 1) {
        g0 += __shfl_xor(g0, off, 32);
        g1 += __shfl_xor(g1, off, 32);
    }
    if (lane == 0) {
        g0 += bg[0];
        g1 += bg[1];
        const float m  = fmaxf(g0, g1);
        const float e0 = __expf(g0 - m), e1 = __expf(g1 - m);
        const float inv = 1.f / (e0 + e1);
        gate[bt * 2 + 0] = e0 * inv;
        gate[bt * 2 + 1] = e1 * inv;
    }
}

extern "C" void kernel_launch(void* const* d_in, const int* in_sizes, int n_in,
                              void* d_out, int out_size, void* d_ws, size_t ws_size,
                              hipStream_t stream)
{
    (void)in_sizes; (void)n_in; (void)out_size; (void)ws_size;

    const float*         source = (const float*)d_in[0];         // [B,S,ES]
    const float*         target = (const float*)d_in[1];         // [B,T,ET]
    const unsigned char* mask   = (const unsigned char*)d_in[2]; // [B,S] bool
    const float*         Ws     = (const float*)d_in[3];         // [H,ES]
    const float*         Wt     = (const float*)d_in[4];         // [H,ET]
    const float*         Wr     = (const float*)d_in[5];         // [H]
    const float*         br     = (const float*)d_in[6];         // scalar
    const float*         Wg     = (const float*)d_in[7];         // [2,ET]
    const float*         bg     = (const float*)d_in[8];         // [2]

    float* out  = (float*)d_out;
    float* cp   = out;                        // [B,T,S]
    float* gate = out + (size_t)B_ * T_ * S_; // [B,T,2]

    // Workspace layout (all power-of-two sizes, naturally aligned):
    float*    sL   = (float*)d_ws;                        // [B*S, H]  2 MB f32
    float*    tL   = sL + (size_t)B_ * S_ * H_;           // [B*T, H]  1 MB f32
    _Float16* srcH = (_Float16*)(tL + (size_t)B_ * T_ * H_); // 512 KB
    _Float16* tgtH = srcH + (size_t)B_ * S_ * ES_;           // 256 KB
    _Float16* WsH  = tgtH + (size_t)B_ * T_ * ES_;           // 1 MB
    _Float16* WtH  = WsH  + (size_t)H_ * ES_;                // 1 MB

    // One-shot f32 -> f16 operand conversion (bandwidth-bound pre-pass)
    const int nSrc8 = (B_ * S_ * ES_) / 8;   // 32768
    const int nTgt8 = (B_ * T_ * ES_) / 8;   // 16384
    const int nW8   = (H_ * ES_) / 8;        // 65536
    cvt_f32_f16<<<dim3(nSrc8 / 256), dim3(256), 0, stream>>>(source, srcH, nSrc8);
    cvt_f32_f16<<<dim3(nTgt8 / 256), dim3(256), 0, stream>>>(target, tgtH, nTgt8);
    cvt_f32_f16<<<dim3(nW8 / 256),   dim3(256), 0, stream>>>(Ws, WsH, nW8);
    cvt_f32_f16<<<dim3(nW8 / 256),   dim3(256), 0, stream>>>(Wt, WtH, nW8);

    // Projections via WMMA (one wave / 16x64 strip, 4 accumulators)
    gemm_nt_wmma_f16<<<dim3(H_ / 64, (B_ * S_) / 16), dim3(32), 0, stream>>>(srcH, WsH, sL);
    gemm_nt_wmma_f16<<<dim3(H_ / 64, (B_ * T_) / 16), dim3(32), 0, stream>>>(tgtH, WtH, tL);

    // Fused tanh-attention reduction
    attn_tanh_reduce<<<dim3(B_ * T_), dim3(256), 0, stream>>>(sL, tL, Wr, br, mask, cp);

    // Gate softmax
    gate_softmax<<<dim3(B_ * T_), dim3(32), 0, stream>>>(target, Wg, bg, gate);
}
